// Coupling_10033043603801
// MI455X (gfx1250) — matrix operand order repaired and checked by reading, not compiled, not run on hardware
//
#include <hip/hip_runtime.h>
#include <math.h>

// ---------------------------------------------------------------------------
// RealNVP coupling layer for MI455X (gfx1250, wave32, WMMA).
// Compute-bound (~240 GFLOP vs ~220MB HBM): bf16 WMMA 16x16x32, f32 accum.
// Global->LDS via CDNA5 async-load path (ASYNCcnt), 3-deep pipeline with
// partial s_wait_asynccnt thresholds; tail fully peeled for a branch-free
// steady-state loop.
// ---------------------------------------------------------------------------

typedef __bf16 bf16;
typedef __attribute__((ext_vector_type(4)))  __bf16 v4bf;
typedef __attribute__((ext_vector_type(8)))  __bf16 v8bf;
typedef __attribute__((ext_vector_type(16))) __bf16 v16bf;
typedef __attribute__((ext_vector_type(2)))  float  v2f;
typedef __attribute__((ext_vector_type(4)))  float  v4f;
typedef __attribute__((ext_vector_type(8)))  float  v8f;

#define B_ROWS 16384
#define DIM_   1024
#define HID_   2048
#define HALF_  512

#define TILE_M 128
#define TILE_N 128
#define TILE_K 32
#define LDS_LD (TILE_K + 8)   // 40 bf16 = 20 dwords: conflict-free frag reads
#define NBUF   3              // 3-deep async pipeline

__device__ __forceinline__ v16bf join16(v8bf lo, v8bf hi) {
  return __builtin_shufflevector(lo, hi, 0,1,2,3,4,5,6,7,8,9,10,11,12,13,14,15);
}

__device__ __forceinline__ float gelu_exact(float x) {
  return 0.5f * x * (1.0f + erff(x * 0.70710678118654752f));
}

// generic LDS pointer -> 32-bit LDS offset (generic = {aperture, off32})
__device__ __forceinline__ unsigned lds_off32(const void* p) {
  return (unsigned)(unsigned long long)(uintptr_t)p;
}

// CDNA5 direct global->LDS async copy, tracked by ASYNCcnt
__device__ __forceinline__ void async_ld_b128(unsigned ldsoff, const void* g) {
  asm volatile("global_load_async_to_lds_b128 %0, %1, off"
               :: "v"(ldsoff), "v"((unsigned long long)(uintptr_t)g)
               : "memory");
}
// ASYNCcnt completes in order: <=4 outstanding <=> oldest slab (4 loads) landed
__device__ __forceinline__ void wait_async_le4() {
  asm volatile("s_wait_asynccnt 0x4" ::: "memory");
}
__device__ __forceinline__ void wait_async0() {
  asm volatile("s_wait_asynccnt 0x0" ::: "memory");
}

// ---------------------------------------------------------------------------
// f32 -> bf16 bulk convert, 8 elements/thread (b128 in, b128 out)
// ---------------------------------------------------------------------------
__global__ void cvt_f32_bf16_v8(const float* __restrict__ in,
                                bf16* __restrict__ out, size_t n8) {
  size_t i = (size_t)blockIdx.x * blockDim.x + threadIdx.x;
  size_t stride = (size_t)gridDim.x * blockDim.x;
  for (; i < n8; i += stride) {
    const v4f a = ((const v4f*)in)[2 * i];
    const v4f b = ((const v4f*)in)[2 * i + 1];
    v8bf o;
    o[0]=(bf16)a[0]; o[1]=(bf16)a[1]; o[2]=(bf16)a[2]; o[3]=(bf16)a[3];
    o[4]=(bf16)b[0]; o[5]=(bf16)b[1]; o[6]=(bf16)b[2]; o[7]=(bf16)b[3];
    ((v8bf*)out)[i] = o;
  }
}

// zm[m] = (bf16) z[2m]  (even columns == flat stride-2), 4 outputs/thread
__global__ void extract_even_bf16_v4(const float* __restrict__ z,
                                     bf16* __restrict__ zm, size_t n4) {
  size_t i = (size_t)blockIdx.x * blockDim.x + threadIdx.x;
  size_t stride = (size_t)gridDim.x * blockDim.x;
  for (; i < n4; i += stride) {
    const v4f a = ((const v4f*)z)[2 * i];
    const v4f b = ((const v4f*)z)[2 * i + 1];
    v4bf o;
    o[0]=(bf16)a[0]; o[1]=(bf16)a[2]; o[2]=(bf16)b[0]; o[3]=(bf16)b[2];
    ((v4bf*)zm)[i] = o;
  }
}

// ---------------------------------------------------------------------------
// C = epilogue(A @ B^T + bias)
//   A: MxK bf16 row-major, B: NxK bf16 row-major (weight layout [N][K])
//   mode 0: gelu -> bf16 out (ld = N);  mode 1: plain -> f32 out (ld = N)
// 256 threads = 8 wave32; wave tile 32x64 = 2x4 wmma accumulators.
// 3-buffer LDS fed by global_load_async_to_lds_b128; peeled drain steps.
// ---------------------------------------------------------------------------
__global__ __launch_bounds__(256) void gemm_nt_bf16(
    const bf16* __restrict__ A, const bf16* __restrict__ B,
    const float* __restrict__ bias, void* __restrict__ Cout,
    int M, int N, int K, int mode)
{
  __shared__ bf16 As[NBUF][TILE_M * LDS_LD];
  __shared__ bf16 Bs[NBUF][TILE_N * LDS_LD];

  const int tid  = threadIdx.x;
  const int lane = tid & 31;
  const int wid  = tid >> 5;
  const int wm   = (wid & 3) * 32;   // wave row offset in tile
  const int wn   = (wid >> 2) * 64;  // wave col offset in tile

  const int m0 = blockIdx.y * TILE_M;
  const int n0 = blockIdx.x * TILE_N;

  // loader: 2 threads per row, 32B (2 x b128) each -> 128x32 bf16 tile
  const int lrow = tid >> 1;
  const int lcol = (tid & 1) * 16;

  const int half = lane >> 4;  // K-half select (ISA 7.12.2 layouts)
  const int l15  = lane & 15;

  const bf16* agBase = A + (size_t)(m0 + lrow) * K + lcol;
  const bf16* bgBase = B + (size_t)(n0 + lrow) * K + lcol;
  const unsigned asOff0 = lds_off32(&As[0][lrow * LDS_LD + lcol]);
  const unsigned bsOff0 = lds_off32(&Bs[0][lrow * LDS_LD + lcol]);
  const unsigned bufStride = lds_off32(&As[1][0]) - lds_off32(&As[0][0]);

  v8f acc[2][4];
  #pragma unroll
  for (int i = 0; i < 2; ++i)
    #pragma unroll
    for (int j = 0; j < 4; ++j) acc[i][j] = (v8f)0.0f;

  // issue global->LDS async loads for K-slab k0 into buffer `buf`
  auto issue_tile = [&](int k0, int buf) {
    const unsigned bofs = (unsigned)buf * bufStride;
    async_ld_b128(asOff0 + bofs,      agBase + k0);
    async_ld_b128(asOff0 + bofs + 16, agBase + k0 + 8);
    async_ld_b128(bsOff0 + bofs,      bgBase + k0);
    async_ld_b128(bsOff0 + bofs + 16, bgBase + k0 + 8);
  };

  // one K-slab of MMA work from LDS buffer `bc`
  auto compute_slab = [&](int bc) {
    const bf16* asb = &As[bc][0];
    const bf16* bsb = &Bs[bc][0];
    v16bf afrag[2], bfrag[4];
    // A 16x32 frag: lane holds row (l&15); halves K {0..7,16..23}/{8..15,24..31}
    #pragma unroll
    for (int i = 0; i < 2; ++i) {
      const bf16* p = asb + (wm + i * 16 + l15) * LDS_LD + half * 8;
      afrag[i] = join16(*(const v8bf*)p, *(const v8bf*)(p + 16));
    }
    // B 32x16 frag: lane holds col (l&15); halves K {0..7,16..23}/{8..15,24..31}
    #pragma unroll
    for (int j = 0; j < 4; ++j) {
      const bf16* p = bsb + (wn + j * 16 + l15) * LDS_LD + half * 8;
      bfrag[j] = join16(*(const v8bf*)p, *(const v8bf*)(p + 16));
    }
    #pragma unroll
    for (int i = 0; i < 2; ++i)
      #pragma unroll
      for (int j = 0; j < 4; ++j)
        acc[i][j] = __builtin_amdgcn_wmma_f32_16x16x32_bf16(
            false, afrag[i], false, bfrag[j], (short)0, acc[i][j],
            false, false);
  };

  const int nsteps = K / TILE_K;   // >= 16 for all layers
  issue_tile(0, 0);
  issue_tile(TILE_K, 1);

  // steady state: branch-free wait/barrier/issue/compute
  int bc = 0;          // buffer being computed
  int bi = 2;          // buffer receiving slab step+2
  for (int step = 0; step < nsteps - 2; ++step) {
    wait_async_le4();  // in-order ASYNCcnt: oldest slab (4 loads) landed
    __syncthreads();   // all waves' loads visible; prior reads of bi done
    issue_tile((step + 2) * TILE_K, bi);
    compute_slab(bc);
    bc = (bc == NBUF - 1) ? 0 : bc + 1;
    bi = (bi == NBUF - 1) ? 0 : bi + 1;
  }
  // drain step nsteps-2: one slab still in flight behind it
  wait_async_le4();
  __syncthreads();
  compute_slab(bc);
  bc = (bc == NBUF - 1) ? 0 : bc + 1;
  // drain step nsteps-1: nothing newer in flight -> full wait
  wait_async0();
  __syncthreads();
  compute_slab(bc);

  // epilogue: C/D layout -> row = half*8 + r, col = l&15
  #pragma unroll
  for (int i = 0; i < 2; ++i) {
    #pragma unroll
    for (int j = 0; j < 4; ++j) {
      const int col = n0 + wn + j * 16 + l15;
      const float bv = bias[col];
      #pragma unroll
      for (int r = 0; r < 8; ++r) {
        const int row = m0 + wm + i * 16 + half * 8 + r;
        float v = acc[i][j][r] + bv;
        if (mode == 0) {
          ((bf16*)Cout)[(size_t)row * N + col] = (bf16)gelu_exact(v);
        } else {
          ((float*)Cout)[(size_t)row * N + col] = v;
        }
      }
    }
  }
}

// ---------------------------------------------------------------------------
// finalize: z_out[:,2j]=z[:,2j]; z_out[:,2j+1]=z[:,2j+1]*exp(s_j)+t_j;
//           log_det_out = log_det + sum_j s_j.   One block per row.
// ---------------------------------------------------------------------------
__global__ __launch_bounds__(256) void finalize_kernel(
    const float* __restrict__ z, const float* __restrict__ log_det,
    const float* __restrict__ st, float* __restrict__ z_out,
    float* __restrict__ logdet_out)
{
  __shared__ float red[8];
  const int row = blockIdx.x;
  const int tid = threadIdx.x;

  const float* stp = st + (size_t)row * DIM_;
  const v2f*   zp  = (const v2f*)(z + (size_t)row * DIM_);
  v2f*         zo  = (v2f*)(z_out + (size_t)row * DIM_);

  float sum = 0.0f;
  for (int j = tid; j < HALF_; j += 256) {
    const float s = stp[j];
    const float t = stp[HALF_ + j];
    const v2f  zv = zp[j];               // {z_even, z_odd}
    v2f ov;
    ov[0] = zv[0];
    ov[1] = zv[1] * expf(s) + t;
    zo[j] = ov;
    sum += s;
  }
  // wave32 shuffle reduce, then cross-wave via LDS
  #pragma unroll
  for (int off = 16; off; off >>= 1) sum += __shfl_down(sum, off, 32);
  if ((tid & 31) == 0) red[tid >> 5] = sum;
  __syncthreads();
  if (tid == 0) {
    float tot = 0.0f;
    #pragma unroll
    for (int w = 0; w < 8; ++w) tot += red[w];
    logdet_out[row] = log_det[row] + tot;
  }
}

// ---------------------------------------------------------------------------
extern "C" void kernel_launch(void* const* d_in, const int* in_sizes, int n_in,
                              void* d_out, int out_size, void* d_ws, size_t ws_size,
                              hipStream_t stream) {
  const float* z       = (const float*)d_in[0];  // 16384x1024
  const float* log_det = (const float*)d_in[1];  // 16384
  const float* W1      = (const float*)d_in[2];  // 2048x512
  const float* b1      = (const float*)d_in[3];  // 2048
  const float* W2      = (const float*)d_in[4];  // 2048x2048
  const float* b2      = (const float*)d_in[5];  // 2048
  const float* W3      = (const float*)d_in[6];  // 1024x2048
  const float* b3      = (const float*)d_in[7];  // 1024

  float* z_out      = (float*)d_out;                         // 16384x1024
  float* logdet_out = (float*)d_out + (size_t)B_ROWS * DIM_; // 16384

  // workspace carve-out
  char* ws = (char*)d_ws;
  size_t off = 0;
  bf16* zm  = (bf16*)(ws + off); off += (size_t)B_ROWS * HALF_ * sizeof(bf16);
  bf16* W1b = (bf16*)(ws + off); off += (size_t)HID_ * HALF_ * sizeof(bf16);
  bf16* W2b = (bf16*)(ws + off); off += (size_t)HID_ * HID_ * sizeof(bf16);
  bf16* W3b = (bf16*)(ws + off); off += (size_t)DIM_ * HID_ * sizeof(bf16);
  bf16* h1  = (bf16*)(ws + off); off += (size_t)B_ROWS * HID_ * sizeof(bf16);
  bf16* h2  = (bf16*)(ws + off); off += (size_t)B_ROWS * HID_ * sizeof(bf16);
  float* st = (float*)(ws + off); off += (size_t)B_ROWS * DIM_ * sizeof(float);
  (void)ws_size; (void)in_sizes; (void)n_in; (void)out_size;

  // 1) precision conversion (bf16 inputs feed the WMMA pipes)
  extract_even_bf16_v4<<<2048, 256, 0, stream>>>(
      z, zm, (size_t)B_ROWS * HALF_ / 4);
  cvt_f32_bf16_v8<<<512, 256, 0, stream>>>(W1, W1b, (size_t)HID_ * HALF_ / 8);
  cvt_f32_bf16_v8<<<2048, 256, 0, stream>>>(W2, W2b, (size_t)HID_ * HID_ / 8);
  cvt_f32_bf16_v8<<<1024, 256, 0, stream>>>(W3, W3b, (size_t)DIM_ * HID_ / 8);

  // 2) three WMMA GEMMs (fused bias + exact GELU where needed)
  dim3 blk(256);
  dim3 g1(HID_ / TILE_N, B_ROWS / TILE_M);   // 16 x 128
  gemm_nt_bf16<<<g1, blk, 0, stream>>>(zm, W1b, b1, h1,
                                       B_ROWS, HID_, HALF_, /*mode=*/0);
  dim3 g2(HID_ / TILE_N, B_ROWS / TILE_M);   // 16 x 128
  gemm_nt_bf16<<<g2, blk, 0, stream>>>(h1, W2b, b2, h2,
                                       B_ROWS, HID_, HID_, /*mode=*/0);
  dim3 g3(DIM_ / TILE_N, B_ROWS / TILE_M);   // 8 x 128
  gemm_nt_bf16<<<g3, blk, 0, stream>>>(h2, W3b, b3, st,
                                       B_ROWS, DIM_, HID_, /*mode=*/1);

  // 3) affine coupling epilogue + per-row log-det reduction
  finalize_kernel<<<B_ROWS, 256, 0, stream>>>(z, log_det, st, z_out, logdet_out);
}
